// Architecture_23338852287020
// MI455X (gfx1250) — matrix operand order repaired and checked by reading, not compiled, not run on hardware
//
#include <hip/hip_runtime.h>
#include <cstdint>
#include <cstddef>

// ---- problem dims ----
#define B_   16
#define T_   8
#define C_   3
#define H_   14
#define W_   14
#define HW_  196
#define HID_ 32
#define IND_ 16
#define NN_  3
#define KIN0 588          // C*HW
#define KP0  608          // KIN0 padded to multiple of 32 (19*32)
#define KHID 6272         // HID*HW (= 196*32, multiple of 32)
#define NBU  3136         // IND*HW
#define NTD  9408         // (IND+HID)*HW

typedef __attribute__((ext_vector_type(16))) __bf16 v16bf;
typedef __attribute__((ext_vector_type(8)))  float  v8f;

__device__ __forceinline__ unsigned short f2bf(float f) {
  union { float f; unsigned u; } x; x.f = f;
  unsigned r = x.u + 0x7FFFu + ((x.u >> 16) & 1u);   // round-to-nearest-even
  return (unsigned short)(r >> 16);
}

// ---------------------------------------------------------------------------
// WMMA GEMM: out(16 x Nvalid, f32) = A(16 x K, bf16) * W^T + bias
//   W is row-major (Nrows x K) bf16, one row per output feature.
//   One wave per 16-wide N tile; K marched in 64-chunks with 2 accumulators.
//   Fragment layouts per CDNA5 ISA 7.12.2 (16-bit A 16x32, B 32x16, f32 C/D).
// ---------------------------------------------------------------------------
__global__ void gemm16_wmma(const unsigned short* __restrict__ A,
                            const unsigned short* __restrict__ Wt,
                            const float* __restrict__ bias,
                            float* __restrict__ out,
                            int K, int ntiles, int Nvalid)
{
  const int lane = threadIdx.x & 31;
  const int wave = threadIdx.x >> 5;
  const int tile = blockIdx.x * (blockDim.x >> 5) + wave;
  if (tile >= ntiles) return;                 // wave-uniform: EXEC stays all-1s
  const int n0   = tile * 16;
  const int col  = lane & 15;
  const int half = lane >> 4;

  int orow = n0 + col; if (orow >= Nvalid) orow = Nvalid - 1;  // clamp OOB rows
  const unsigned short* arow = A  + (size_t)col  * K;          // A row m = col
  const unsigned short* wrow = Wt + (size_t)orow * K;          // B col = out feature

  union Frag { v16bf v; uint4 q[2]; };
  v8f acc0, acc1;
  #pragma unroll
  for (int r = 0; r < 8; ++r) { acc0[r] = 0.f; acc1[r] = 0.f; }

  int k = 0;
  for (; k + 64 <= K; k += 64) {
    Frag a0, b0, a1, b1;
    // A 16x32 layout: lanes 0-15 hold K [k..k+8) in v0..3, [k+16..k+24) in v4..7;
    // lanes 16-31 shifted by 8.
    a0.q[0] = *(const uint4*)(arow + k      + half * 8);
    a0.q[1] = *(const uint4*)(arow + k + 16 + half * 8);
    // B 32x16 layout: same K order as A (lanes 0-15 [k..k+8),[k+16..k+24); lanes 16-31 shifted by 8).
    b0.q[0] = *(const uint4*)(wrow + k      + half * 8);
    b0.q[1] = *(const uint4*)(wrow + k + 16 + half * 8);
    a1.q[0] = *(const uint4*)(arow + k + 32 + half * 8);
    a1.q[1] = *(const uint4*)(arow + k + 48 + half * 8);
    b1.q[0] = *(const uint4*)(wrow + k + 32 + half * 8);
    b1.q[1] = *(const uint4*)(wrow + k + 48 + half * 8);
    acc0 = __builtin_amdgcn_wmma_f32_16x16x32_bf16(false, a0.v, false, b0.v,
                                                   (short)0, acc0, false, false);
    acc1 = __builtin_amdgcn_wmma_f32_16x16x32_bf16(false, a1.v, false, b1.v,
                                                   (short)0, acc1, false, false);
  }
  for (; k < K; k += 32) {                    // 32-wide tail (K=608 case)
    Frag a0, b0;
    a0.q[0] = *(const uint4*)(arow + k      + half * 8);
    a0.q[1] = *(const uint4*)(arow + k + 16 + half * 8);
    b0.q[0] = *(const uint4*)(wrow + k      + half * 8);
    b0.q[1] = *(const uint4*)(wrow + k + 16 + half * 8);
    acc0 = __builtin_amdgcn_wmma_f32_16x16x32_bf16(false, a0.v, false, b0.v,
                                                   (short)0, acc0, false, false);
  }
  #pragma unroll
  for (int r = 0; r < 8; ++r) acc0[r] += acc1[r];

  const int oc = n0 + col;
  if (oc < Nvalid) {
    const float bia   = bias[oc];
    const int   mbase = half * 8;             // D layout: lanes 0-15 M=r, 16-31 M=r+8
    #pragma unroll
    for (int r = 0; r < 8; ++r)
      out[(size_t)(mbase + r) * Nvalid + oc] = acc0[r] + bia;
  }
}

// ---- f32 -> bf16 weight conversion (grid-stride) ----
__global__ void cvt_f32_bf16(const float* __restrict__ in,
                             unsigned short* __restrict__ out, size_t n) {
  size_t i = (size_t)blockIdx.x * blockDim.x + threadIdx.x;
  size_t st = (size_t)gridDim.x * blockDim.x;
  for (; i < n; i += st) out[i] = f2bf(in[i]);
}

// ---- f32 -> bf16 with per-row zero padding (bu_w0: 588 -> 608) ----
__global__ void cvt_pad_bf16(const float* __restrict__ in,
                             unsigned short* __restrict__ out,
                             int rows, int kin, int kout) {
  int i = blockIdx.x * blockDim.x + threadIdx.x;
  if (i >= rows * kout) return;
  int r = i / kout, k = i % kout;
  out[i] = (k < kin) ? f2bf(in[(size_t)r * kin + k]) : (unsigned short)0;
}

// ---- x[:, t] -> bf16 activation, zero-padded to 16 x 608 ----
__global__ void x_to_bf16(const float* __restrict__ x, int t,
                          unsigned short* __restrict__ out) {
  int i = blockIdx.x * blockDim.x + threadIdx.x;
  if (i >= B_ * KP0) return;
  int b = i / KP0, k = i % KP0;
  out[i] = (k < KIN0) ? f2bf(x[((size_t)b * T_ + t) * KIN0 + k]) : (unsigned short)0;
}

// ---- fused maxpool3x3(stride1,pad1) + f32->bf16 flatten: (16,32,14,14)->(16,6272)
__global__ void pool3_bf16(const float* __restrict__ h,
                           unsigned short* __restrict__ out) {
  int i = blockIdx.x * blockDim.x + threadIdx.x;
  if (i >= B_ * HID_ * HW_) return;
  int pos = i % HW_, rest = i / HW_;
  int y = pos / W_, xx = pos % W_;
  const float* base = h + (size_t)rest * HW_;
  float m = -3.4e38f;
  #pragma unroll
  for (int dy = -1; dy <= 1; ++dy) {
    int yy = y + dy; if (yy < 0 || yy >= H_) continue;
    #pragma unroll
    for (int dx = -1; dx <= 1; ++dx) {
      int xc = xx + dx; if (xc < 0 || xc >= W_) continue;
      m = fmaxf(m, base[yy * W_ + xc]);
    }
  }
  out[i] = f2bf(m);
}

// ---- relu + f32->bf16 ----
__global__ void relu_bf16(const float* __restrict__ in,
                          unsigned short* __restrict__ out, int n) {
  int i = blockIdx.x * blockDim.x + threadIdx.x;
  if (i < n) out[i] = f2bf(fmaxf(in[i], 0.f));
}

// ---------------------------------------------------------------------------
// GRU gate conv (48ch -> 64ch, 3x3 SAME) + sigmoid.  LDS-tiled per (batch,
// 16-oc group).  Writes z and r*h.
// ---------------------------------------------------------------------------
__global__ void cell_gates(const float* __restrict__ bu, const float* __restrict__ h,
                           const float* __restrict__ td, int hasTd,
                           const float* __restrict__ Wg, const float* __restrict__ bg,
                           float* __restrict__ z, float* __restrict__ rh)
{
  __shared__ float comb[48 * HW_];     // 37.6 KB
  __shared__ float wsm[16 * 48 * 9];   // 27.6 KB
  const int b = blockIdx.x >> 2;
  const int g = blockIdx.x & 3;        // oc group: [16g, 16g+16)
  for (int i = threadIdx.x; i < 48 * HW_; i += blockDim.x) {
    int ic = i / HW_, pos = i % HW_;
    float v = (ic < IND_) ? bu[((size_t)b * IND_ + ic) * HW_ + pos]
                          : h [((size_t)b * HID_ + (ic - IND_)) * HW_ + pos];
    if (hasTd) v += td[((size_t)b * 48 + ic) * HW_ + pos];
    comb[i] = v;
  }
  for (int i = threadIdx.x; i < 16 * 48 * 9; i += blockDim.x)
    wsm[i] = Wg[(size_t)(g * 16) * 48 * 9 + i];
  __syncthreads();

  for (int i = threadIdx.x; i < 16 * HW_; i += blockDim.x) {
    int ocl = i / HW_, pos = i % HW_;
    int y = pos / W_, xx = pos % W_;
    float acc = bg[g * 16 + ocl];
    for (int ic = 0; ic < 48; ++ic) {
      const float* cm = comb + ic * HW_;
      const float* wp = wsm + (ocl * 48 + ic) * 9;
      #pragma unroll
      for (int ky = 0; ky < 3; ++ky) {
        int yy = y + ky - 1; if (yy < 0 || yy >= H_) continue;
        #pragma unroll
        for (int kx = 0; kx < 3; ++kx) {
          int xc = xx + kx - 1; if (xc < 0 || xc >= W_) continue;
          acc += cm[yy * W_ + xc] * wp[ky * 3 + kx];
        }
      }
    }
    float s = 1.f / (1.f + __expf(-acc));
    int oc = g * 16 + ocl;
    if (oc < HID_)                          // r gate -> store r*h
      rh[((size_t)b * HID_ + oc) * HW_ + pos] =
          s * h[((size_t)b * HID_ + oc) * HW_ + pos];
    else                                    // z gate
      z[((size_t)b * HID_ + (oc - HID_)) * HW_ + pos] = s;
  }
}

// ---------------------------------------------------------------------------
// GRU candidate conv (48ch -> 32ch) + tanh + state update h = (1-z)h + z*cand
// ---------------------------------------------------------------------------
__global__ void cell_cand(const float* __restrict__ bu, const float* __restrict__ rh,
                          const float* __restrict__ z, float* __restrict__ h,
                          const float* __restrict__ Wc, const float* __restrict__ bc)
{
  __shared__ float comb[48 * HW_];
  __shared__ float wsm[16 * 48 * 9];
  const int b = blockIdx.x >> 1;
  const int g = blockIdx.x & 1;        // oc group: [16g, 16g+16)
  for (int i = threadIdx.x; i < 48 * HW_; i += blockDim.x) {
    int ic = i / HW_, pos = i % HW_;
    comb[i] = (ic < IND_) ? bu[((size_t)b * IND_ + ic) * HW_ + pos]
                          : rh[((size_t)b * HID_ + (ic - IND_)) * HW_ + pos];
  }
  for (int i = threadIdx.x; i < 16 * 48 * 9; i += blockDim.x)
    wsm[i] = Wc[(size_t)(g * 16) * 48 * 9 + i];
  __syncthreads();

  for (int i = threadIdx.x; i < 16 * HW_; i += blockDim.x) {
    int ocl = i / HW_, pos = i % HW_;
    int y = pos / W_, xx = pos % W_;
    float acc = bc[g * 16 + ocl];
    for (int ic = 0; ic < 48; ++ic) {
      const float* cm = comb + ic * HW_;
      const float* wp = wsm + (ocl * 48 + ic) * 9;
      #pragma unroll
      for (int ky = 0; ky < 3; ++ky) {
        int yy = y + ky - 1; if (yy < 0 || yy >= H_) continue;
        #pragma unroll
        for (int kx = 0; kx < 3; ++kx) {
          int xc = xx + kx - 1; if (xc < 0 || xc >= W_) continue;
          acc += cm[yy * W_ + xc] * wp[ky * 3 + kx];
        }
      }
    }
    int oc = g * 16 + ocl;
    size_t idx = ((size_t)b * HID_ + oc) * HW_ + pos;
    float cand = tanhf(acc);
    float zz = z[idx], hv = h[idx];
    h[idx] = (1.f - zz) * hv + zz * cand;
  }
}

// ---- fc2 head: out(16,10) = relu(p1(16,100)) @ fc2_w^T + fc2_b ----
__global__ void fc2_head(const float* __restrict__ p1, const float* __restrict__ w,
                         const float* __restrict__ bias, float* __restrict__ out) {
  int i = threadIdx.x;
  if (i >= B_ * 10) return;
  int b = i / 10, j = i % 10;
  float acc = bias[j];
  for (int k = 0; k < 100; ++k)
    acc += fmaxf(p1[b * 100 + k], 0.f) * w[j * 100 + k];
  out[i] = acc;
}

// ---------------------------------------------------------------------------
extern "C" void kernel_launch(void* const* d_in, const int* in_sizes, int n_in,
                              void* d_out, int out_size, void* d_ws, size_t ws_size,
                              hipStream_t stream)
{
  const float* x     = (const float*)d_in[0];
  const float* Wg    = (const float*)d_in[1];
  const float* bg    = (const float*)d_in[2];
  const float* Wc    = (const float*)d_in[3];
  const float* bc    = (const float*)d_in[4];
  const float* bu_w0 = (const float*)d_in[5];
  const float* bu_b0 = (const float*)d_in[6];
  const float* bu_w1 = (const float*)d_in[7];
  const float* bu_b1 = (const float*)d_in[8];
  const float* bu_w2 = (const float*)d_in[9];
  const float* bu_b2 = (const float*)d_in[10];
  const float* td_w0 = (const float*)d_in[11];
  const float* td_b0 = (const float*)d_in[12];
  const float* td_w1 = (const float*)d_in[13];
  const float* td_b1 = (const float*)d_in[14];
  const float* fc1_w = (const float*)d_in[15];
  const float* fc1_b = (const float*)d_in[16];
  const float* fc2_w = (const float*)d_in[17];
  const float* fc2_b = (const float*)d_in[18];
  float* out = (float*)d_out;

  // ---- workspace carve-out (~324 MB; assumes ws_size is sufficient) ----
  char* wsp = (char*)d_ws;
  size_t off = 0;
  auto carve = [&](size_t bytes) -> void* {
    void* p = wsp + off;
    off = (off + bytes + 255) & ~(size_t)255;
    return p;
  };
  unsigned short* wbu0 = (unsigned short*)carve((size_t)NBU * KP0 * 2);
  unsigned short* wbu1 = (unsigned short*)carve((size_t)NBU * KHID * 2);
  unsigned short* wbu2 = (unsigned short*)carve((size_t)NBU * KHID * 2);
  unsigned short* wtd0 = (unsigned short*)carve((size_t)NTD * KHID * 2);
  unsigned short* wtd1 = (unsigned short*)carve((size_t)NTD * KHID * 2);
  unsigned short* wfc1 = (unsigned short*)carve((size_t)100 * KHID * 2);
  float* hidall = (float*)carve((size_t)NN_ * B_ * HID_ * HW_ * 4);
  unsigned short* actbuf = (unsigned short*)carve((size_t)B_ * KHID * 2);
  float* bubuf = (float*)carve((size_t)B_ * NBU * 4);
  float* tdbuf = (float*)carve((size_t)B_ * NTD * 4);
  float* zbuf  = (float*)carve((size_t)B_ * HID_ * HW_ * 4);
  float* rhbuf = (float*)carve((size_t)B_ * HID_ * HW_ * 4);
  float* p1    = (float*)carve((size_t)B_ * 100 * 4);

  float* hid[NN_] = { hidall,
                      hidall + (size_t)B_ * HID_ * HW_,
                      hidall + (size_t)2 * B_ * HID_ * HW_ };

  const int CT = 256;
  // ---- one-time (per launch) bf16 weight conversion ----
  cvt_pad_bf16<<<(NBU * KP0 + CT - 1) / CT, CT, 0, stream>>>(bu_w0, wbu0, NBU, KIN0, KP0);
  cvt_f32_bf16<<<4096, CT, 0, stream>>>(bu_w1, wbu1, (size_t)NBU * KHID);
  cvt_f32_bf16<<<4096, CT, 0, stream>>>(bu_w2, wbu2, (size_t)NBU * KHID);
  cvt_f32_bf16<<<8192, CT, 0, stream>>>(td_w0, wtd0, (size_t)NTD * KHID);
  cvt_f32_bf16<<<8192, CT, 0, stream>>>(td_w1, wtd1, (size_t)NTD * KHID);
  cvt_f32_bf16<<<1024, CT, 0, stream>>>(fc1_w, wfc1, (size_t)100 * KHID);

  // ---- zero hidden state (capture-safe) ----
  hipMemsetAsync(hidall, 0, (size_t)NN_ * B_ * HID_ * HW_ * 4, stream);

  const int poolGrid = (B_ * HID_ * HW_ + CT - 1) / CT;
  // ---- unrolled time loop (T + N - 1 = 10 steps, sequential node updates) ----
  for (int t = 0; t < T_ + NN_ - 1; ++t) {
    for (int node = 0; node < NN_; ++node) {
      if (node == 0) { if (t >= T_) continue; }     // input exhausted
      else           { if (t == 0) continue; }      // upstream hidden all-zero
      // bottom-up path -> bubuf (16 x 3136)
      if (node == 0) {
        x_to_bf16<<<(B_ * KP0 + CT - 1) / CT, CT, 0, stream>>>(x, t, actbuf);
        gemm16_wmma<<<49, 128, 0, stream>>>(actbuf, wbu0, bu_b0, bubuf, KP0, 196, NBU);
      } else {
        pool3_bf16<<<poolGrid, CT, 0, stream>>>(hid[node - 1], actbuf);
        gemm16_wmma<<<49, 128, 0, stream>>>(actbuf,
            (node == 1) ? wbu1 : wbu2, (node == 1) ? bu_b1 : bu_b2,
            bubuf, KHID, 196, NBU);
      }
      // top-down path -> tdbuf (16 x 9408)
      const int hasTd = (node < NN_ - 1 && t >= 2) ? 1 : 0;
      if (hasTd) {
        pool3_bf16<<<poolGrid, CT, 0, stream>>>(hid[node + 1], actbuf);
        gemm16_wmma<<<147, 128, 0, stream>>>(actbuf,
            (node == 0) ? wtd0 : wtd1, (node == 0) ? td_b0 : td_b1,
            tdbuf, KHID, 588, NTD);
      }
      // ConvGRU cell
      cell_gates<<<64, 256, 0, stream>>>(bubuf, hid[node], tdbuf, hasTd,
          Wg + (size_t)node * 64 * 48 * 9, bg + node * 64, zbuf, rhbuf);
      cell_cand<<<32, 256, 0, stream>>>(bubuf, rhbuf, zbuf, hid[node],
          Wc + (size_t)node * 32 * 48 * 9, bc + node * 32);
    }
  }

  // ---- classifier head ----
  relu_bf16<<<(B_ * KHID + CT - 1) / CT, CT, 0, stream>>>(hid[NN_ - 1], actbuf, B_ * KHID);
  gemm16_wmma<<<2, 128, 0, stream>>>(actbuf, wfc1, fc1_b, p1, KHID, 7, 100);
  fc2_head<<<1, 192, 0, stream>>>(p1, fc2_w, fc2_b, out);
}
